// FusionLayer_10393820856636
// MI455X (gfx1250) — compile-verified, matrix operand order checked by reading, not run on hardware
//
#include <hip/hip_runtime.h>
#include <hip/hip_bf16.h>
#include <math.h>

// ---------------------------------------------------------------------------
// Types for CDNA5 WMMA (wave32, 16x16x32 bf16 -> f32 accum)
// ---------------------------------------------------------------------------
typedef unsigned short u16;
typedef __attribute__((ext_vector_type(16))) __bf16 v16bf;
typedef __attribute__((ext_vector_type(8)))  float  v8f;
typedef __attribute__((ext_vector_type(4)))  unsigned int u32x4;
typedef __attribute__((ext_vector_type(8)))  unsigned int u32x8;

union Frag {
    v16bf v;
    uint4 q[2];   // two 16B chunks = 16 bf16 values per lane
};

__device__ __forceinline__ v8f zero8() {
    v8f z = {0.f, 0.f, 0.f, 0.f, 0.f, 0.f, 0.f, 0.f};
    return z;
}

// round-to-nearest-even f32 -> bf16 (stored as u16)
__device__ __forceinline__ u16 f2bf(float f) {
    union { float f; unsigned int u; } c; c.f = f;
    unsigned int u = c.u;
    unsigned int r = u + 0x7FFFu + ((u >> 16) & 1u);
    return (u16)(r >> 16);
}

__device__ __forceinline__ v8f wmma_bf16(const Frag& a, const Frag& b, v8f c) {
    // v_wmma_f32_16x16x32_bf16  (neg_a, A, neg_b, B, c_mod, C, reuse_a, reuse_b)
    return __builtin_amdgcn_wmma_f32_16x16x32_bf16(false, a.v, false, b.v,
                                                   (short)0, c, false, false);
}

// Problem dims
#define Bsz   32
#define Ssz   512
#define Hsz   768
#define NHh   12
#define HDd   64
#define MROWS (Bsz * Ssz)   // 16384
#define TPAD  36            // table rows padded per batch (0..32 valid, 33..35 zero)

// ---------------------------------------------------------------------------
// Prep kernels
// ---------------------------------------------------------------------------
__global__ void f32_to_bf16_kern(const float* __restrict__ x, u16* __restrict__ y, int n) {
    int i = blockIdx.x * 256 + threadIdx.x;
    if (i < n) y[i] = f2bf(x[i]);
}

// table[b][c][j]: c==0 -> 0 (gather null row); c in 1..32 -> colemb masked by category
__global__ void build_table_kern(const float* __restrict__ ce, const int* __restrict__ cat,
                                 u16* __restrict__ tb) {
    int idx = blockIdx.x * 256 + threadIdx.x;
    const int n = Bsz * TPAD * Hsz;
    if (idx >= n) return;
    int j = idx % Hsz;
    int c = (idx / Hsz) % TPAD;
    int b = idx / (Hsz * TPAD);
    float v = 0.f;
    if (c >= 1 && c <= 32) {
        int col = c - 1;
        if (cat[b * 32 + col] == 1) v = ce[((size_t)b * 32 + col) * Hsz + j];
    }
    tb[idx] = f2bf(v);
}

// ---------------------------------------------------------------------------
// Tensor Data Mover: DMA one [rows x 32] bf16 tile (row stride 768 elems) from
// global into LDS, padding each 32-elem row to 40 elems (LDT) in LDS.
//   pad_interval = 3  -> pad after every 16 DWORDs (= 32 bf16 = one tile row)
//   pad_amount   = 3  -> insert 4 DWORDs (= 8 bf16)  => LDS row stride 40 bf16
// Issued once per wave (EXEC ignored); tracked by TENSORcnt.
// ---------------------------------------------------------------------------
#define LDT 40   // LDS row stride (u16): 80B, 16B-aligned chunks, bank-spread

__device__ __forceinline__ void tdm_load_tile(const u16* gsrc, unsigned lds_byte_off,
                                              unsigned tensor_rows)
{
    unsigned long long ga = (unsigned long long)(size_t)gsrc;
    u32x4 g0;
    g0[0] = 1u;                                          // count=1, no gather
    g0[1] = lds_byte_off;                                // lds_addr
    g0[2] = (unsigned)(ga & 0xFFFFFFFFu);                // global_addr[31:0]
    g0[3] = (unsigned)((ga >> 32) & 0x01FFFFFFu)         // global_addr[56:32]
          | 0x80000000u;                                 // type=2 (bits 127:126)
    u32x8 g1;
    g1[0] = (1u << 16)       // data_size = 2 bytes
          | (1u << 20)       // pad_enable
          | (3u << 22)       // pad_interval: 16 DWORDs
          | (3u << 25);      // pad_amount:   4 DWORDs
    g1[1] = (768u & 0xFFFFu) << 16;                      // tensor_dim0 = 768 (lo16)
    g1[2] = (tensor_rows & 0xFFFFu) << 16;               // dim0 hi=0 | tensor_dim1 lo16
    g1[3] = (tensor_rows >> 16) | (32u << 16);           // tensor_dim1 hi | tile_dim0=32
    g1[4] = 128u;                                        // tile_dim1=128, tile_dim2=0
    g1[5] = 768u;                                        // tensor_dim0_stride[31:0]
    g1[6] = 0u;                                          // stride0 hi | stride1 lo
    g1[7] = 0u;                                          // stride1 hi
    asm volatile("tensor_load_to_lds %0, %1" :: "s"(g0), "s"(g1) : "memory");
}

__device__ __forceinline__ unsigned lds_off_of(const void* p) {
    return (unsigned)(unsigned long long)(size_t)p;      // low 32 bits = LDS offset
}

// ---------------------------------------------------------------------------
// Generic bf16 WMMA GEMM:  out[row,col] = sum_k A[row,k] * Bw[col,k]  (+ epilogue)
// Tile: 128x128x32, 8 waves (4 along M x 2 along N), wave region 32x64.
// LDS tiles are double-buffered and filled by TDM (wave 0 issues, TENSORcnt<=2
// guarantees the tile being consumed is resident while the next streams in).
//   EPI 0: P-proj:  outF = acc + bias
//   EPI 1: h:       g = gelu(acc + bias + P[b, vm[row], col]); store f32 + bf16
//   EPI 2: qkv:     outB[((b*12+h)*512+s)*64+d] = bf16(acc + bias)
//   EPI 3: y:       outF = acc + bias + resid
// ---------------------------------------------------------------------------
template <int EPI>
__global__ __launch_bounds__(256) void gemm_kern(
    const u16* __restrict__ A, const u16* __restrict__ Bw,
    const float* __restrict__ bias,
    float* __restrict__ outF, u16* __restrict__ outB,
    const float* __restrict__ Pg, const int* __restrict__ vm,
    const float* __restrict__ resid, unsigned Mrows)
{
    const int K = Hsz;
    __shared__ u16 As[2][128 * LDT];
    __shared__ u16 Bs[2][128 * LDT];

    int tid   = threadIdx.x;
    int lane  = tid & 31;
    int wave  = tid >> 5;
    int waveM = wave & 3;    // 4 wave-rows of 32
    int waveN = wave >> 2;   // 2 wave-cols of 64
    int m0 = blockIdx.x * 128;
    int n0 = blockIdx.y * 128;
    int lr = lane & 15;
    int hf = lane >> 4;

    v8f acc[2][4];
    for (int i = 0; i < 2; i++)
        for (int j = 0; j < 4; j++) acc[i][j] = zero8();

    // prologue: TDM-stage K-tile 0 into buffer 0
    if (wave == 0) {
        tdm_load_tile(A  + (size_t)m0 * K, lds_off_of(&As[0][0]), Mrows);
        tdm_load_tile(Bw + (size_t)n0 * K, lds_off_of(&Bs[0][0]), Hsz);
    }

    const int NK = K / 32;   // 24
    for (int ki = 0; ki < NK; ki++) {
        int buf = ki & 1;
        if (wave == 0) {
            if (ki + 1 < NK) {
                int k1 = (ki + 1) * 32;
                tdm_load_tile(A  + (size_t)m0 * K + k1, lds_off_of(&As[buf ^ 1][0]), Mrows);
                tdm_load_tile(Bw + (size_t)n0 * K + k1, lds_off_of(&Bs[buf ^ 1][0]), Hsz);
                __builtin_amdgcn_s_wait_tensorcnt((short)2);  // tile ki resident
            } else {
                __builtin_amdgcn_s_wait_tensorcnt((short)0);
            }
        }
        __syncthreads();     // tile ki visible to all waves

        const u16* as = &As[buf][0];
        const u16* bs = &Bs[buf][0];
        Frag a[2], b[4];
        for (int mb = 0; mb < 2; mb++) {
            int row = waveM * 32 + mb * 16 + lr;
            a[mb].q[0] = *(const uint4*)&as[row * LDT + hf * 8];
            a[mb].q[1] = *(const uint4*)&as[row * LDT + 16 + hf * 8];
        }
        for (int nb = 0; nb < 4; nb++) {
            int col = waveN * 64 + nb * 16 + lr;
            b[nb].q[0] = *(const uint4*)&bs[col * LDT + hf * 8];
            b[nb].q[1] = *(const uint4*)&bs[col * LDT + 16 + hf * 8];
        }
        for (int mb = 0; mb < 2; mb++)
            for (int nb = 0; nb < 4; nb++)
                acc[mb][nb] = wmma_bf16(a[mb], b[nb], acc[mb][nb]);
        __syncthreads();     // consumption done: buf may be overwritten next iter
    }

    // epilogue: D frag element e -> (row = base + 8*hf + e, col = base + lr)
    for (int mb = 0; mb < 2; mb++) {
        int rowb = m0 + waveM * 32 + mb * 16 + hf * 8;
        for (int nb = 0; nb < 4; nb++) {
            int col   = n0 + waveN * 64 + nb * 16 + lr;
            float bcl = bias[col];
            for (int e = 0; e < 8; e++) {
                int row = rowb + e;
                float val = acc[mb][nb][e] + bcl;
                if (EPI == 0) {
                    outF[(size_t)row * Hsz + col] = val;
                } else if (EPI == 1) {
                    int bIdx = row >> 9;
                    int c    = vm[row];
                    val += Pg[((size_t)bIdx * TPAD + c) * Hsz + col];
                    float g = 0.5f * val * (1.f + erff(val * 0.70710678118f));
                    outF[(size_t)row * Hsz + col] = g;
                    outB[(size_t)row * Hsz + col] = f2bf(g);
                } else if (EPI == 2) {
                    int bIdx = row >> 9, s = row & 511;
                    int hh = col >> 6, d = col & 63;
                    outB[(((size_t)bIdx * NHh + hh) * Ssz + s) * HDd + d] = f2bf(val);
                } else {
                    val += resid[(size_t)row * Hsz + col];
                    outF[(size_t)row * Hsz + col] = val;
                }
            }
        }
    }
}

// ---------------------------------------------------------------------------
// Flash-style attention. Grid: (B*NH, S/128). Block: 256 (8 waves).
// Wave w owns query rows [w*16, w*16+16). Streams K/V in tiles of 64.
// Q,K,V layout: [B*NH][S][64] bf16. ctx: [B][S][768] bf16.
// ---------------------------------------------------------------------------
__global__ __launch_bounds__(256) void attn_kern(
    const u16* __restrict__ Q, const u16* __restrict__ Kk,
    const u16* __restrict__ V, u16* __restrict__ ctx)
{
    __shared__ u16   Ks[64 * 72];
    __shared__ u16   Vt[64 * 72];          // transposed V: [d][t]
    __shared__ float Sf[8][16][68];        // per-wave f32 scores; reused in-place as bf16 probs
    __shared__ float alphaS[8][16];
    __shared__ float linvS[8][16];

    int bh   = blockIdx.x;                 // 0..383
    int q0   = blockIdx.y * 128;
    int b    = bh / NHh, head = bh % NHh;
    int tid  = threadIdx.x;
    int lane = tid & 31;
    int wave = tid >> 5;
    int lr   = lane & 15;
    int hf   = lane >> 4;

    // Q fragments: 16 rows x K=64 -> 2 frags, loaded straight from global
    Frag aq[2];
    {
        int qrow = q0 + wave * 16 + lr;
        const u16* qp = Q + ((size_t)bh * Ssz + qrow) * HDd;
        for (int kk = 0; kk < 2; kk++) {
            aq[kk].q[0] = *(const uint4*)(qp + kk * 32 + hf * 8);
            aq[kk].q[1] = *(const uint4*)(qp + kk * 32 + 16 + hf * 8);
        }
    }

    v8f o[4];
    for (int nb = 0; nb < 4; nb++) o[nb] = zero8();
    float m_i = -1e30f, l_i = 0.f;

    for (int t0 = 0; t0 < Ssz; t0 += 64) {
        __syncthreads();
        // stage K tile [64][64]
        for (int c = tid; c < 64 * 8; c += 256) {
            int r = c >> 3, kc = (c & 7) * 8;
            *(uint4*)&Ks[r * 72 + kc] =
                *(const uint4*)&Kk[((size_t)bh * Ssz + t0 + r) * HDd + kc];
        }
        // stage V tile transposed: Vt[d][t]
        for (int c = tid; c < 64 * 64; c += 256) {
            int t = c >> 6, d = c & 63;
            Vt[d * 72 + t] = V[((size_t)bh * Ssz + t0 + t) * HDd + d];
        }
        __syncthreads();

        // scores: 16x64 per wave, scaled by 1/sqrt(64)
        for (int nb = 0; nb < 4; nb++) {
            v8f sc = zero8();
            int ck = nb * 16 + lr;
            for (int kk = 0; kk < 2; kk++) {
                Frag bk;
                bk.q[0] = *(const uint4*)&Ks[ck * 72 + kk * 32 + hf * 8];
                bk.q[1] = *(const uint4*)&Ks[ck * 72 + kk * 32 + 16 + hf * 8];
                sc = wmma_bf16(aq[kk], bk, sc);
            }
            for (int e = 0; e < 8; e++)
                Sf[wave][e + 8 * hf][nb * 16 + lr] = sc[e] * 0.125f;
        }
        __syncthreads();

        // online softmax, one lane per query row; compact exp() in place as bf16
        if (lane < 16) {
            int r = lr;
            float rmax = -1e30f;
            for (int c2 = 0; c2 < 64; c2++) rmax = fmaxf(rmax, Sf[wave][r][c2]);
            float mnew = fmaxf(m_i, rmax);
            float a    = __expf(m_i - mnew);
            float sum  = 0.f;
            u16* prow = (u16*)&Sf[wave][r][0];
            for (int c2 = 0; c2 < 64; c2++) {
                float e = __expf(Sf[wave][r][c2] - mnew);
                prow[c2] = f2bf(e);           // write (2B) trails read (4B): safe
                sum += e;
            }
            l_i = l_i * a + sum;
            m_i = mnew;
            alphaS[wave][r] = a;
        }
        __syncthreads();

        // rescale running O, then O += P * V
        float af[8];
        for (int e = 0; e < 8; e++) af[e] = alphaS[wave][e + 8 * hf];
        for (int nb = 0; nb < 4; nb++)
            for (int e = 0; e < 8; e++) o[nb][e] *= af[e];

        const u16* prow = (const u16*)&Sf[wave][lr][0];
        for (int kk = 0; kk < 2; kk++) {
            Frag pa;
            pa.q[0] = *(const uint4*)(prow + kk * 32 + hf * 8);
            pa.q[1] = *(const uint4*)(prow + kk * 32 + 16 + hf * 8);
            for (int nb = 0; nb < 4; nb++) {
                Frag bv;
                int d = nb * 16 + lr;
                bv.q[0] = *(const uint4*)&Vt[d * 72 + kk * 32 + hf * 8];
                bv.q[1] = *(const uint4*)&Vt[d * 72 + kk * 32 + 16 + hf * 8];
                o[nb] = wmma_bf16(pa, bv, o[nb]);
            }
        }
    }

    if (lane < 16) linvS[wave][lr] = 1.f / l_i;
    __syncthreads();
    float li[8];
    for (int e = 0; e < 8; e++) li[e] = linvS[wave][e + 8 * hf];
    for (int nb = 0; nb < 4; nb++) {
        for (int e = 0; e < 8; e++) {
            int s   = q0 + wave * 16 + e + 8 * hf;
            int col = head * HDd + nb * 16 + lr;
            ctx[((size_t)b * Ssz + s) * Hsz + col] = f2bf(o[nb][e] * li[e]);
        }
    }
}

// ---------------------------------------------------------------------------
// Row LayerNorm: out = (y - mu) * rsqrt(var + 1e-12) * g + b
// ---------------------------------------------------------------------------
__global__ __launch_bounds__(256) void ln_kern(const float* __restrict__ y,
                                               const float* __restrict__ g,
                                               const float* __restrict__ b,
                                               float* __restrict__ out)
{
    __shared__ float red[256];
    int row = blockIdx.x, tid = threadIdx.x;
    const float* yr = y + (size_t)row * Hsz;

    float s = 0.f;
    for (int j = tid; j < Hsz; j += 256) s += yr[j];
    red[tid] = s; __syncthreads();
    for (int st = 128; st > 0; st >>= 1) {
        if (tid < st) red[tid] += red[tid + st];
        __syncthreads();
    }
    float mu = red[0] * (1.f / Hsz);
    __syncthreads();

    float v = 0.f;
    for (int j = tid; j < Hsz; j += 256) { float d = yr[j] - mu; v += d * d; }
    red[tid] = v; __syncthreads();
    for (int st = 128; st > 0; st >>= 1) {
        if (tid < st) red[tid] += red[tid + st];
        __syncthreads();
    }
    float var = red[0] * (1.f / Hsz);
    float rs  = rsqrtf(var + 1e-12f);
    for (int j = tid; j < Hsz; j += 256)
        out[(size_t)row * Hsz + j] = (yr[j] - mu) * rs * g[j] + b[j];
}

// ---------------------------------------------------------------------------
// Host launcher
// ---------------------------------------------------------------------------
extern "C" void kernel_launch(void* const* d_in, const int* in_sizes, int n_in,
                              void* d_out, int out_size, void* d_ws, size_t ws_size,
                              hipStream_t stream)
{
    const float* hidden = (const float*)d_in[0];
    const float* colemb = (const float*)d_in[1];
    const int*   colcat = (const int*)  d_in[2];
    const int*   vmask  = (const int*)  d_in[3];
    const float* Wd = (const float*)d_in[4];   const float* bd = (const float*)d_in[5];
    const float* Wc = (const float*)d_in[6];   const float* bc = (const float*)d_in[7];
    const float* Wq = (const float*)d_in[8];   const float* bq = (const float*)d_in[9];
    const float* Wk = (const float*)d_in[10];  const float* bk = (const float*)d_in[11];
    const float* Wv = (const float*)d_in[12];  const float* bv = (const float*)d_in[13];
    const float* Wo = (const float*)d_in[14];  const float* bo = (const float*)d_in[15];
    const float* lng = (const float*)d_in[16];
    const float* lnb = (const float*)d_in[17];
    float* out = (float*)d_out;

    char* ws = (char*)d_ws;
    size_t off = 0;
    auto carve = [&](size_t bytes) -> char* {
        char* p = ws + off;
        off = (off + bytes + 255) & ~(size_t)255;
        return p;
    };

    const size_t NE = (size_t)MROWS * Hsz;   // 16384*768
    const size_t NW = (size_t)Hsz * Hsz;

    u16*   Xb   = (u16*)  carve(NE * 2);
    u16*   Wdb  = (u16*)  carve(NW * 2);
    u16*   Wcb  = (u16*)  carve(NW * 2);
    u16*   Wqb  = (u16*)  carve(NW * 2);
    u16*   Wkb  = (u16*)  carve(NW * 2);
    u16*   Wvb  = (u16*)  carve(NW * 2);
    u16*   Wob  = (u16*)  carve(NW * 2);
    u16*   Tb   = (u16*)  carve((size_t)Bsz * TPAD * Hsz * 2);
    float* P    = (float*)carve((size_t)Bsz * TPAD * Hsz * 4);
    float* hF   = (float*)carve(NE * 4);
    u16*   hB   = (u16*)  carve(NE * 2);
    u16*   qB   = (u16*)  carve(NE * 2);
    u16*   kB   = (u16*)  carve(NE * 2);
    u16*   vB   = (u16*)  carve(NE * 2);
    u16*   ctxB = (u16*)  carve(NE * 2);
    float* yF   = (float*)carve(NE * 4);
    (void)ws_size; (void)in_sizes; (void)n_in; (void)out_size;

    // --- precision prep (f32 -> bf16) ---
    {
        int n = (int)NE;
        f32_to_bf16_kern<<<(n + 255) / 256, 256, 0, stream>>>(hidden, Xb, n);
        int nw = (int)NW;
        f32_to_bf16_kern<<<(nw + 255) / 256, 256, 0, stream>>>(Wd, Wdb, nw);
        f32_to_bf16_kern<<<(nw + 255) / 256, 256, 0, stream>>>(Wc, Wcb, nw);
        f32_to_bf16_kern<<<(nw + 255) / 256, 256, 0, stream>>>(Wq, Wqb, nw);
        f32_to_bf16_kern<<<(nw + 255) / 256, 256, 0, stream>>>(Wk, Wkb, nw);
        f32_to_bf16_kern<<<(nw + 255) / 256, 256, 0, stream>>>(Wv, Wvb, nw);
        f32_to_bf16_kern<<<(nw + 255) / 256, 256, 0, stream>>>(Wo, Wob, nw);
        int nt = Bsz * TPAD * Hsz;
        build_table_kern<<<(nt + 255) / 256, 256, 0, stream>>>(colemb, colcat, Tb);
    }

    // --- projected gather table: P[b,c,:] = table[b,c,:] @ Wc^T + bc  (M=1152) ---
    {
        dim3 g((Bsz * TPAD) / 128, Hsz / 128);
        gemm_kern<0><<<g, 256, 0, stream>>>(Tb, Wcb, bc, P, nullptr, nullptr, nullptr,
                                            nullptr, (unsigned)(Bsz * TPAD));
    }

    dim3 gM(MROWS / 128, Hsz / 128);

    // --- h = gelu(X @ Wd^T + bd + P[b, vm]) -> f32 + bf16 ---
    gemm_kern<1><<<gM, 256, 0, stream>>>(Xb, Wdb, bd, hF, hB, P, vmask, nullptr, MROWS);

    // --- Q,K,V projections (bf16, head-transposed layout [B,NH,S,64]) ---
    gemm_kern<2><<<gM, 256, 0, stream>>>(hB, Wqb, bq, nullptr, qB, nullptr, nullptr, nullptr, MROWS);
    gemm_kern<2><<<gM, 256, 0, stream>>>(hB, Wkb, bk, nullptr, kB, nullptr, nullptr, nullptr, MROWS);
    gemm_kern<2><<<gM, 256, 0, stream>>>(hB, Wvb, bv, nullptr, vB, nullptr, nullptr, nullptr, MROWS);

    // --- attention -> ctx bf16 [B,S,768] ---
    {
        dim3 g(Bsz * NHh, Ssz / 128);
        attn_kern<<<g, 256, 0, stream>>>(qB, kB, vB, ctxB);
    }

    // --- y = ctx @ Wo^T + bo + h ---
    gemm_kern<3><<<gM, 256, 0, stream>>>(ctxB, Wob, bo, yF, nullptr, nullptr, nullptr, hF, MROWS);

    // --- out = LN(y) ---
    ln_kern<<<MROWS, 256, 0, stream>>>(yF, lng, lnb, out);
}
